// SuperResolNet_33114197852634
// MI455X (gfx1250) — matrix operand order repaired and checked by reading, not compiled, not run on hardware
//
#include <hip/hip_runtime.h>

// ---------------------------------------------------------------------------
// CDNA5 (gfx1250) SuperResolNet forward.
// Heavy convs -> implicit GEMM with V_WMMA_F32_16X16X32_F16 (f32 accum).
// Per-tap packed-weight chunks staged into LDS via Tensor Data Mover
// (tensor_load_to_lds + s_wait_tensorcnt), double-buffered.
// ---------------------------------------------------------------------------

typedef __attribute__((ext_vector_type(16))) _Float16     v16h;
typedef __attribute__((ext_vector_type(8)))  float        v8f;
typedef __attribute__((ext_vector_type(4)))  unsigned int u32x4;
typedef __attribute__((ext_vector_type(8)))  int          i32x8;
typedef __attribute__((ext_vector_type(4)))  int          i32x4;

#ifndef __has_builtin
#define __has_builtin(x) 0
#endif

#if __has_builtin(__builtin_amdgcn_tensor_load_to_lds)
#define USE_TDM 1
#if __has_include(<hip/amd_detail/amd_gfx1250_TDM.h>)
#define TDM_ARGS6 1
#else
#define TDM_ARGS6 0
#endif
#else
#define USE_TDM 0
#endif

#if __has_builtin(__builtin_amdgcn_s_wait_tensorcnt)
#define HAVE_WTC 1
#else
#define HAVE_WTC 0
#endif

struct Q32 { u32x4 a, b; };   // 32 bytes == one v16h fragment

static __device__ __forceinline__ v16h frag_from(const _Float16* p0,
                                                 const _Float16* p1) {
  Q32 t;
  t.a = *(const u32x4*)p0;
  t.b = *(const u32x4*)p1;
  return __builtin_bit_cast(v16h, t);
}

static __device__ __forceinline__ v16h frag_zero() {
  Q32 t;
  u32x4 z = {0u, 0u, 0u, 0u};
  t.a = z; t.b = z;
  return __builtin_bit_cast(v16h, t);
}

// ---------------------------------------------------------------------------
// TDM 1-D staging: global -> LDS, `bytes` must be a multiple of 8.
// D# layout per CDNA5 ISA:
//   g0: [1:0]=count=1, [63:32]=lds_addr, [120:64]=global_addr, [127:126]=type=2
//   g1: [17:16]=data_size=3 (8B), [79:48]=tensor_dim0, [127:112]=tile_dim0,
//       [207:160]=tensor_dim0_stride; all other dims 0 (1-D copy).
// ---------------------------------------------------------------------------
static __device__ __forceinline__ void stage_issue(_Float16* lds_dst,
                                                   const _Float16* gsrc,
                                                   int bytes) {
#if USE_TDM
  if (threadIdx.x == 0) {
    unsigned long long ga = (unsigned long long)(const void*)gsrc;
    unsigned ldsOff = (unsigned)(unsigned long long)(void*)lds_dst;  // LDS byte offset (low 32b of flat addr)
    unsigned units = (unsigned)bytes >> 3;  // 8-byte elements
    u32x4 g0;
    g0.x = 1u;                                                   // count=1
    g0.y = ldsOff;                                               // lds_addr
    g0.z = (unsigned)ga;                                         // global_addr[31:0]
    g0.w = (unsigned)((ga >> 32) & 0x01ffffffu) | (2u << 30);    // addr[56:32] | type=2
    i32x8 g1;
    g1[0] = (int)(3u << 16);                   // data_size = 3 (8 bytes)
    g1[1] = (int)((units & 0xffffu) << 16);    // tensor_dim0[15:0]  @bits63:48
    g1[2] = (int)((units >> 16) & 0xffffu);    // tensor_dim0[31:16] @bits79:64
    g1[3] = (int)((units & 0xffffu) << 16);    // tile_dim0          @bits127:112
    g1[4] = 0;                                 // tile_dim1/2 = 0
    g1[5] = (int)units;                        // tensor_dim0_stride[31:0]
    g1[6] = 0;
    g1[7] = 0;
    i32x4 gz = {0, 0, 0, 0};
#if TDM_ARGS6
    i32x8 gz8 = {0, 0, 0, 0, 0, 0, 0, 0};
    __builtin_amdgcn_tensor_load_to_lds(g0, g1, gz, gz, gz8, 0);
#else
    __builtin_amdgcn_tensor_load_to_lds(g0, g1, gz, gz, 0);
#endif
  }
#else
  // fallback: cooperative copy (whole block)
  const u32x4* s = (const u32x4*)gsrc;
  u32x4* d = (u32x4*)lds_dst;
  const int n16 = bytes >> 4;
  for (int i = threadIdx.x; i < n16; i += 256) d[i] = s[i];
#endif
}

static __device__ __forceinline__ void stage_finish() {
#if USE_TDM && HAVE_WTC
  __builtin_amdgcn_s_wait_tensorcnt(0);
#endif
  __syncthreads();
}

// ---------------------------------------------------------------------------
// Weight packing: OIDHW f32 -> per-lane B-fragment f16.
// Linear order: [(t*KB+kb)*NB + nb] * 512 + lane*16 + j
// B layout (32x16 f16): lane<16: N=lane,  K=j ; lane>=16: N=lane-16, K=16+j
// ---------------------------------------------------------------------------
__global__ void pack_w(const float* __restrict__ w, _Float16* __restrict__ wp,
                       int Cin, int Cpad, int Cout, int KD) {
  const int KB = Cpad >> 5, NB = Cout >> 4;
  const long total = (long)KD * 9 * KB * NB * 512;
  long idx = (long)blockIdx.x * blockDim.x + threadIdx.x;
  if (idx >= total) return;
  int j    = (int)(idx & 15);
  int lane = (int)((idx >> 4) & 31);
  long r   = idx >> 9;
  int nb = (int)(r % NB); r /= NB;
  int kb = (int)(r % KB); r /= KB;
  int t  = (int)r;
  int kd = t / 9, ky = (t % 9) / 3, kx = t % 3;
  int n   = nb * 16 + (lane & 15);
  int kl  = ((lane >> 4) & 1) * 16 + j;
  int cin = kb * 32 + kl;
  float v = (cin < Cin) ? w[(((long)(n * Cin + cin) * KD + kd) * 3 + ky) * 3 + kx]
                        : 0.f;
  wp[idx] = (_Float16)v;
}

// ---------------------------------------------------------------------------
// Implicit-GEMM conv (KD x 3 x 3, pad=1 in H/W, valid in D), NB = Cout/16.
// Per-tap weight chunk (KB*NB*512 halves <= 16384) TDM-staged into LDS,
// double buffered.  Each wave computes a 16(M) x Cout tile.
// Output: f32 channels-last [pixel][Cout].
// ---------------------------------------------------------------------------
template <int NB>
__global__ void __launch_bounds__(256)
conv_wmma_t(const _Float16* __restrict__ act, const _Float16* __restrict__ wp,
            const float* __restrict__ bias, float* __restrict__ out,
            int N, int D, int H, int W, int Cpad,
            int KD, int stride, int zeropad,
            int Dout, int Hout, int Wout, int Cout) {
  __shared__ _Float16 wsm[2 * 16384];          // 64 KB: two weight-chunk buffers

  const int lane = threadIdx.x & 31;
  const int wv   = threadIdx.x >> 5;
  const int KB   = Cpad >> 5;
  const int chunk = KB * NB * 512;             // halves per tap
  const int taps  = KD * 9;
  const long Mtot = (long)N * Dout * Hout * Wout;
  const long tile = (long)blockIdx.x * 8 + wv;

  long m = tile * 16 + (lane & 15);
  if (m >= Mtot) m = Mtot - 1;                 // clamp loads; stores predicated
  long r = m;
  const int ow = (int)(r % Wout); r /= Wout;
  const int oh = (int)(r % Hout); r /= Hout;
  const int od = (int)(r % Dout);
  const int n  = (int)(r / Dout);

  const v8f vzero = {0.f, 0.f, 0.f, 0.f, 0.f, 0.f, 0.f, 0.f};
  v8f acc[NB];
#pragma unroll
  for (int i = 0; i < NB; ++i) acc[i] = vzero;

  const int aoff = (lane & 16) ? 16 : 0;       // A-fragment K-half select

  stage_issue(wsm, wp, chunk * 2);
  stage_finish();

  for (int t = 0; t < taps; ++t) {
    if (t + 1 < taps)
      stage_issue(wsm + ((t + 1) & 1) * 16384, wp + (long)(t + 1) * chunk,
                  chunk * 2);

    const _Float16* wbuf = wsm + (t & 1) * 16384;
    const int kd = t / 9, ky = (t % 9) / 3, kx = t % 3;
    const int id = od + kd;                    // valid conv over D
    const int ih = oh * stride + ky - 1;
    const int iw = ow * stride + kx - 1;
    const bool oob = (ih < 0) || (ih >= H) || (iw < 0) || (iw >= W);
    const int ch = ih < 0 ? 0 : (ih >= H ? H - 1 : ih);
    const int cw = iw < 0 ? 0 : (iw >= W ? W - 1 : iw);
    const _Float16* rowp =
        act + ((((long)n * D + id) * H + ch) * W + cw) * Cpad;
    const bool kill = (zeropad != 0) && oob;
    __builtin_prefetch(rowp + Cpad, 0, 1);     // global_prefetch_b8

    for (int kb = 0; kb < KB; ++kb) {
      v16h afrag;
      if (kill) {
        afrag = frag_zero();
      } else {
        const _Float16* p = rowp + kb * 32 + aoff;
        afrag = frag_from(p, p + 8);
      }
      const _Float16* wtap = wbuf + kb * NB * 512;
#pragma unroll
      for (int nb = 0; nb < NB; ++nb) {
        const _Float16* q = wtap + (nb * 32 + lane) * 16;   // ds_load_b128 x2
        v16h bfrag = frag_from(q, q + 8);
        acc[nb] = __builtin_amdgcn_wmma_f32_16x16x32_f16(
            false, afrag, false, bfrag, (short)0, acc[nb], false, false);
      }
    }
    stage_finish();   // s_wait_tensorcnt 0 + barrier (buffer rotation safe)
  }

  // C/D layout: (vgpr rr, lane l) = M row rr + (l>=16 ? 8 : 0), N = l&15
  const int colb = lane & 15;
  const int mrow = (lane & 16) ? 8 : 0;
#pragma unroll
  for (int rr = 0; rr < 8; ++rr) {
    const long mm = tile * 16 + rr + mrow;
    if (mm < Mtot) {
      float* o = out + mm * Cout;
#pragma unroll
      for (int nb = 0; nb < NB; ++nb)
        o[nb * 16 + colb] = acc[nb][rr] + bias[nb * 16 + colb];
    }
  }
}

// ---------------------------------------------------------------------------
// Instance-norm stats: per (n,c) over DHW, channels-last f32 input.
// ---------------------------------------------------------------------------
__global__ void inorm_stats(const float* __restrict__ x, float* __restrict__ mean,
                            float* __restrict__ rstd, int N, int C, long DHW) {
  const int n = blockIdx.x / C, c = blockIdx.x % C;
  __shared__ float ss[256], sq[256];
  float s = 0.f, q = 0.f;
  for (long i = threadIdx.x; i < DHW; i += blockDim.x) {
    float v = x[(n * DHW + i) * C + c];
    s += v; q += v * v;
  }
  ss[threadIdx.x] = s; sq[threadIdx.x] = q;
  __syncthreads();
  for (int o = 128; o > 0; o >>= 1) {
    if ((int)threadIdx.x < o) {
      ss[threadIdx.x] += ss[threadIdx.x + o];
      sq[threadIdx.x] += sq[threadIdx.x + o];
    }
    __syncthreads();
  }
  if (threadIdx.x == 0) {
    float m = ss[0] / (float)DHW;
    float v = sq[0] / (float)DHW - m * m;
    mean[n * C + c] = m;
    rstd[n * C + c] = rsqrtf(v + 1e-5f);
  }
}

// optional inorm + lrelu, f32 -> f16 channels-last
__global__ void act_apply(const float* __restrict__ x, const float* __restrict__ mean,
                          const float* __restrict__ rstd, _Float16* __restrict__ y,
                          int C, long DHW, int donorm, long total) {
  long idx = (long)blockIdx.x * blockDim.x + threadIdx.x;
  if (idx >= total) return;
  int c  = (int)(idx % C);
  long p = idx / C;
  int n  = (int)(p / DHW);
  float v = x[idx];
  if (donorm) v = (v - mean[n * C + c]) * rstd[n * C + c];
  v = v >= 0.f ? v : 0.01f * v;
  y[idx] = (_Float16)v;
}

// xIm (2,1,9,128,128) f32 -> act [2,9,128,128,32] f16 (c0=data, rest 0)
__global__ void build_x0(const float* __restrict__ x, _Float16* __restrict__ act,
                         long P) {
  long idx = (long)blockIdx.x * blockDim.x + threadIdx.x;
  if (idx >= P * 32) return;
  int c  = (int)(idx & 31);
  long p = idx >> 5;
  act[idx] = (_Float16)(c == 0 ? x[p] : 0.f);
}

// STN input: [16,128,128,128]: c<64 = feat[b,d=0], c>=64 = feat[b,d=l+1]
__global__ void build_stn_in(const _Float16* __restrict__ feat,
                             _Float16* __restrict__ act) {
  long idx = (long)blockIdx.x * blockDim.x + threadIdx.x;
  if (idx >= (long)16 * 128 * 128 * 128) return;
  int c  = (int)(idx & 127);
  int w  = (int)((idx >> 7) & 127);
  int h  = (int)((idx >> 14) & 127);
  int bl = (int)(idx >> 21);
  int b = bl >> 3, l = bl & 7;
  int d = (c < 64) ? 0 : (l + 1);
  int cc = c & 63;
  act[idx] = feat[((((long)b * 9 + d) * 128 + h) * 128 + w) * 64 + cc];
}

// STN head: conv (2,64,3,3) zero-pad + spatial mean + clip -> shift[16][2]
__global__ void stn_head(const _Float16* __restrict__ act, const float* __restrict__ w,
                         const float* __restrict__ b, float* __restrict__ shift,
                         int H, int W, int C, float clipv) {
  const int bl = blockIdx.x;
  __shared__ float s0s[256], s1s[256];
  float a0 = 0.f, a1 = 0.f;
  for (int p = threadIdx.x; p < H * W; p += blockDim.x) {
    int oh = p / W, ow = p % W;
    float acc0 = b[0], acc1 = b[1];
    for (int ky = 0; ky < 3; ++ky) {
      int ih = oh + ky - 1;
      if (ih < 0 || ih >= H) continue;
      for (int kx = 0; kx < 3; ++kx) {
        int iw = ow + kx - 1;
        if (iw < 0 || iw >= W) continue;
        const _Float16* ap = act + (((long)bl * H + ih) * W + iw) * C;
        for (int c = 0; c < C; ++c) {
          float v = (float)ap[c];
          acc0 += v * w[((0 * C + c) * 3 + ky) * 3 + kx];
          acc1 += v * w[((1 * C + c) * 3 + ky) * 3 + kx];
        }
      }
    }
    a0 += acc0; a1 += acc1;
  }
  s0s[threadIdx.x] = a0; s1s[threadIdx.x] = a1;
  __syncthreads();
  for (int o = 128; o > 0; o >>= 1) {
    if ((int)threadIdx.x < o) {
      s0s[threadIdx.x] += s0s[threadIdx.x + o];
      s1s[threadIdx.x] += s1s[threadIdx.x + o];
    }
    __syncthreads();
  }
  if (threadIdx.x == 0) {
    float inv = 1.f / (float)(H * W);
    float v0 = s0s[0] * inv, v1 = s1s[0] * inv;
    v0 = v0 < -clipv ? -clipv : (v0 > clipv ? clipv : v0);
    v1 = v1 < -clipv ? -clipv : (v1 > clipv ? clipv : v1);
    shift[bl * 2 + 0] = v0;
    shift[bl * 2 + 1] = v1;
  }
}

// regFeat d=0 := feat d=0
__global__ void copy_d0(const _Float16* __restrict__ feat,
                        _Float16* __restrict__ regf) {
  long idx = (long)blockIdx.x * blockDim.x + threadIdx.x;
  if (idx >= (long)2 * 128 * 128 * 64) return;
  long per = (long)128 * 128 * 64;
  int b = (int)(idx / per);
  long o = idx % per;
  regf[(long)b * 9 * per + o] = feat[(long)b * 9 * per + o];
}

// bilinear translate-sample of feat[b, l+1] by shift[bl] -> regFeat[b, 1+l]
__global__ void warp_kernel(const _Float16* __restrict__ feat,
                            const float* __restrict__ shift,
                            _Float16* __restrict__ regf) {
  long idx = (long)blockIdx.x * blockDim.x + threadIdx.x;
  if (idx >= (long)16 * 128 * 128 * 64) return;
  const float Wf = 128.f, Hf = 128.f;
  int c  = (int)(idx & 63);
  int w  = (int)((idx >> 6) & 127);
  int h  = (int)((idx >> 13) & 127);
  int bl = (int)(idx >> 20);
  int b = bl >> 3, l = bl & 7;
  float s0 = shift[bl * 2 + 0], s1 = shift[bl * 2 + 1];
  float gx = (2.f * w + 1.f) / Wf - 1.f + s0;
  float gy = (2.f * h + 1.f) / Hf - 1.f + s1;
  float ix = ((gx + 1.f) * Wf - 1.f) * 0.5f;
  float iy = ((gy + 1.f) * Hf - 1.f) * 0.5f;
  float ix0 = floorf(ix), iy0 = floorf(iy);
  float wx1 = ix - ix0, wy1 = iy - iy0;
  const _Float16* img = feat + (((long)b * 9 + (l + 1)) * 128) * 128 * 64;
  float acc = 0.f;
#pragma unroll
  for (int dy = 0; dy < 2; ++dy) {
#pragma unroll
    for (int dx = 0; dx < 2; ++dx) {
      float iyc = iy0 + dy, ixc = ix0 + dx;
      float wgt = (dy ? wy1 : 1.f - wy1) * (dx ? wx1 : 1.f - wx1);
      bool m = (ixc >= 0.f) && (ixc <= Wf - 1.f) && (iyc >= 0.f) && (iyc <= Hf - 1.f);
      float iyp = iyc < 0.f ? 0.f : (iyc > Hf - 1.f ? Hf - 1.f : iyc);
      float ixp = ixc < 0.f ? 0.f : (ixc > Wf - 1.f ? Wf - 1.f : ixc);
      int ii = (int)iyp, jj = (int)ixp;
      float v = (float)img[(((long)ii * 128 + jj) * 64) + c];
      acc += v * wgt * (m ? 1.f : 0.f);
    }
  }
  regf[((((long)b * 9 + (1 + l)) * 128 + h) * 128 + w) * 64 + c] = (_Float16)acc;
}

// Cout=1 conv (1,3,3), zero pad, input act f16 [N,1,H,W,C] -> out f32 [N,H,W]
__global__ void conv_out1(const _Float16* __restrict__ act, const float* __restrict__ w,
                          const float* __restrict__ b, float* __restrict__ out,
                          int N, int H, int W, int C) {
  long idx = (long)blockIdx.x * blockDim.x + threadIdx.x;
  if (idx >= (long)N * H * W) return;
  int ow = (int)(idx % W);
  long r = idx / W;
  int oh = (int)(r % H);
  int n  = (int)(r / H);
  float acc = b[0];
  for (int ky = 0; ky < 3; ++ky) {
    int ih = oh + ky - 1;
    if (ih < 0 || ih >= H) continue;
    for (int kx = 0; kx < 3; ++kx) {
      int iw = ow + kx - 1;
      if (iw < 0 || iw >= W) continue;
      const _Float16* ap = act + (((long)n * H + ih) * W + iw) * C;
      const float* wp2 = w + (ky * 3 + kx);
      for (int c = 0; c < C; ++c) acc += (float)ap[c] * wp2[c * 9];
    }
  }
  out[idx] = acc;
}

// refine input: [2,9,128,128,96]: c<64 regFeat, c==64 hfRes, rest 0
__global__ void build_refine_in(const _Float16* __restrict__ regf,
                                const float* __restrict__ hf,
                                _Float16* __restrict__ act) {
  long idx = (long)blockIdx.x * blockDim.x + threadIdx.x;
  if (idx >= (long)2 * 9 * 128 * 128 * 96) return;
  int c  = (int)(idx % 96);
  long p = idx / 96;
  int w  = (int)(p % 128);
  long r = p / 128;
  int h  = (int)(r % 128);
  r /= 128;
  int b = (int)(r / 9);
  float v;
  if (c < 64)       v = (float)regf[p * 64 + c];
  else if (c == 64) v = hf[((long)b * 128 + h) * 128 + w];
  else              v = 0.f;
  act[idx] = (_Float16)v;
}

// hr = xIm[:,:,0] + hfRes + hfRes2
__global__ void final_out(const float* __restrict__ xIm, const float* __restrict__ hf,
                          const float* __restrict__ hf2, float* __restrict__ out) {
  long idx = (long)blockIdx.x * blockDim.x + threadIdx.x;
  if (idx >= 32768) return;
  int b  = (int)(idx >> 14);
  long hw = idx & 16383;
  out[idx] = xIm[(long)b * 9 * 16384 + hw] + hf[idx] + hf2[idx];
}

// per-(b,dy,dx) window loss: var of diff over (Hc,Wc)
__global__ void lossA(const float* __restrict__ hr, const float* __restrict__ tg,
                      float* __restrict__ ls, int H, int W) {
  const int Hc = H - 12, Wc = W - 12;
  const int blk = blockIdx.x;
  const int b = blk / 169, s = blk % 169, dy = s / 13, dx = s % 13;
  __shared__ float ss[256], sq[256];
  float s1 = 0.f, s2 = 0.f;
  for (int i = threadIdx.x; i < Hc * Wc; i += blockDim.x) {
    int hc = i / Wc, wc = i % Wc;
    float t = tg[((long)b * H + (hc + 6)) * W + (wc + 6)];
    float h = hr[((long)b * H + (hc + dy)) * W + (wc + dx)];
    float d = t - h;
    s1 += d; s2 += d * d;
  }
  ss[threadIdx.x] = s1; sq[threadIdx.x] = s2;
  __syncthreads();
  for (int o = 128; o > 0; o >>= 1) {
    if ((int)threadIdx.x < o) {
      ss[threadIdx.x] += ss[threadIdx.x + o];
      sq[threadIdx.x] += sq[threadIdx.x + o];
    }
    __syncthreads();
  }
  if (threadIdx.x == 0) {
    float M = (float)(Hc * Wc);
    float mean = ss[0] / M;
    ls[blk] = sq[0] / M - mean * mean;
  }
}

__global__ void lossB(const float* __restrict__ ls, float* __restrict__ out) {
  if (threadIdx.x == 0 && blockIdx.x == 0) {
    float acc = 0.f;
    for (int b = 0; b < 2; ++b) {
      float best = 3.4e38f;
      for (int s = 0; s < 169; ++s) {
        float v = ls[b * 169 + s];
        best = v < best ? v : best;
      }
      acc += best;
    }
    out[0] = acc * 0.5f;
  }
}

// ---------------------------------------------------------------------------
// Host orchestration
// ---------------------------------------------------------------------------
extern "C" void kernel_launch(void* const* d_in, const int* in_sizes, int n_in,
                              void* d_out, int out_size, void* d_ws, size_t ws_size,
                              hipStream_t stream) {
  (void)in_sizes; (void)n_in; (void)out_size; (void)ws_size;
  const float* xIm    = (const float*)d_in[0];
  const float* target = (const float*)d_in[1];
  // params flattened with sorted dict keys:
  // fusion(2..9) fusion_out(10,11) primary(12..27) refine(28..35)
  // refine_out(36,37) stn(38..45) stn_out(46,47)
  const float *FW[4], *FB[4], *PW[8], *PB[8], *RW[4], *RB[4], *SW[4], *SB[4];
  for (int i = 0; i < 4; ++i) { FW[i] = (const float*)d_in[2 + 2*i];  FB[i] = (const float*)d_in[3 + 2*i]; }
  const float* FOW = (const float*)d_in[10]; const float* FOB = (const float*)d_in[11];
  for (int i = 0; i < 8; ++i) { PW[i] = (const float*)d_in[12 + 2*i]; PB[i] = (const float*)d_in[13 + 2*i]; }
  for (int i = 0; i < 4; ++i) { RW[i] = (const float*)d_in[28 + 2*i]; RB[i] = (const float*)d_in[29 + 2*i]; }
  const float* ROW = (const float*)d_in[36]; const float* ROB = (const float*)d_in[37];
  for (int i = 0; i < 4; ++i) { SW[i] = (const float*)d_in[38 + 2*i]; SB[i] = (const float*)d_in[39 + 2*i]; }
  const float* SOW = (const float*)d_in[46]; const float* SOB = (const float*)d_in[47];

  char* base = (char*)d_ws;
  auto alloc = [&](size_t bytes) -> void* {
    void* p = (void*)base;
    base += (bytes + 255) & ~(size_t)255;
    return p;
  };
  _Float16* ACT_P = (_Float16*)alloc(33554432ull * 2);  // 16*128*128*128 halves
  _Float16* ACT_Q = (_Float16*)alloc(33554432ull * 2);
  _Float16* FEAT  = (_Float16*)alloc(18874368ull * 2);  // 2*9*128*128*64
  _Float16* REGF  = (_Float16*)alloc(18874368ull * 2);
  float*    CONV  = (float*)alloc(18874368ull * 4);     // raw conv out (f32)
  _Float16* WPK   = (_Float16*)alloc(262144ull * 2);    // packed weights
  float* MEAN   = (float*)alloc(4096 * 4);
  float* RSTD   = (float*)alloc(4096 * 4);
  float* HFRES  = (float*)alloc(32768 * 4);
  float* HFRES2 = (float*)alloc(32768 * 4);
  float* SHIFT  = (float*)alloc(256 * 4);
  float* LOSSS  = (float*)alloc(512 * 4);
  float* hrOut  = (float*)d_out;

  auto cdivl = [](long a, long b) -> unsigned int { return (unsigned int)((a + b - 1) / b); };

  auto conv_layer = [&](const _Float16* ain, _Float16* aout,
                        const float* w, const float* b,
                        int N, int D, int H, int W, int Cin, int Cpad, int Cout,
                        int KD, int stride, int zeropad, int donorm) {
    const int Dout = D - (KD - 1);
    const int Hout = (H - 1) / stride + 1;  // pad=1, k=3
    const int Wout = (W - 1) / stride + 1;
    const int NB = Cout / 16;
    const long packTot = (long)KD * 9 * (Cpad / 32) * NB * 512;
    pack_w<<<dim3(cdivl(packTot, 256)), dim3(256), 0, stream>>>(w, WPK, Cin, Cpad, Cout, KD);
    const long Mtot = (long)N * Dout * Hout * Wout;
    const dim3 grid(cdivl(Mtot, 128)), blk(256);
    if (NB == 8)
      conv_wmma_t<8><<<grid, blk, 0, stream>>>(ain, WPK, b, CONV, N, D, H, W,
                                               Cpad, KD, stride, zeropad,
                                               Dout, Hout, Wout, Cout);
    else
      conv_wmma_t<4><<<grid, blk, 0, stream>>>(ain, WPK, b, CONV, N, D, H, W,
                                               Cpad, KD, stride, zeropad,
                                               Dout, Hout, Wout, Cout);
    const long DHW = (long)Dout * Hout * Wout;
    if (donorm)
      inorm_stats<<<dim3(N * Cout), dim3(256), 0, stream>>>(CONV, MEAN, RSTD, N, Cout, DHW);
    const long tot = Mtot * Cout;
    act_apply<<<dim3(cdivl(tot, 256)), dim3(256), 0, stream>>>(
        CONV, MEAN, RSTD, aout, Cout, DHW, donorm, tot);
  };

  // ---- primary: 8x [conv(1,3,3) rep-pad -> inorm -> lrelu] -----------------
  const long P0 = 2L * 9 * 128 * 128;
  build_x0<<<dim3(cdivl(P0 * 32, 256)), dim3(256), 0, stream>>>(xIm, ACT_P, P0);
  conv_layer(ACT_P, ACT_Q, PW[0], PB[0], 2, 9, 128, 128, 1, 32, 64, 1, 1, 0, 1);
  {
    _Float16* cur = ACT_Q; _Float16* nxt = ACT_P;
    for (int i = 1; i < 8; ++i) {
      _Float16* dst = (i == 7) ? FEAT : nxt;
      conv_layer(cur, dst, PW[i], PB[i], 2, 9, 128, 128, 64, 64, 64, 1, 1, 0, 1);
      nxt = cur; cur = dst;
    }
  }

  // ---- STN: 4x [conv2d zero-pad -> lrelu], head conv + mean + clip ---------
  build_stn_in<<<dim3(cdivl(16L * 128 * 128 * 128, 256)), dim3(256), 0, stream>>>(FEAT, ACT_P);
  conv_layer(ACT_P, ACT_Q, SW[0], SB[0], 16, 1, 128, 128, 128, 128, 128, 1, 2, 1, 0);
  conv_layer(ACT_Q, ACT_P, SW[1], SB[1], 16, 1, 64, 64, 128, 128, 64, 1, 1, 1, 0);
  conv_layer(ACT_P, ACT_Q, SW[2], SB[2], 16, 1, 64, 64, 64, 64, 64, 1, 1, 1, 0);
  conv_layer(ACT_Q, ACT_P, SW[3], SB[3], 16, 1, 64, 64, 64, 64, 64, 1, 1, 1, 0);
  stn_head<<<dim3(16), dim3(256), 0, stream>>>(ACT_P, SOW, SOB, SHIFT, 64, 64, 64, 4.f / 128.f);

  // ---- warp refy by shift -> regFeat ---------------------------------------
  copy_d0<<<dim3(cdivl(2L * 128 * 128 * 64, 256)), dim3(256), 0, stream>>>(FEAT, REGF);
  warp_kernel<<<dim3(cdivl(16L * 128 * 128 * 64, 256)), dim3(256), 0, stream>>>(FEAT, SHIFT, REGF);

  // ---- fusion: 4x [conv3d(3,3,3) rep-pad HW -> inorm -> lrelu], out conv ---
  conv_layer(REGF,  ACT_P, FW[0], FB[0], 2, 9, 128, 128, 64, 64, 64, 3, 1, 0, 1);
  conv_layer(ACT_P, ACT_Q, FW[1], FB[1], 2, 7, 128, 128, 64, 64, 64, 3, 1, 0, 1);
  conv_layer(ACT_Q, ACT_P, FW[2], FB[2], 2, 5, 128, 128, 64, 64, 64, 3, 1, 0, 1);
  conv_layer(ACT_P, ACT_Q, FW[3], FB[3], 2, 3, 128, 128, 64, 64, 64, 3, 1, 0, 1);
  conv_out1<<<dim3(cdivl(2L * 128 * 128, 256)), dim3(256), 0, stream>>>(ACT_Q, FOW, FOB, HFRES, 2, 128, 128, 64);

  // ---- refine: concat(regFeat, hfRes) -> 4x conv3d -> out conv -------------
  build_refine_in<<<dim3(cdivl(2L * 9 * 128 * 128 * 96, 256)), dim3(256), 0, stream>>>(REGF, HFRES, ACT_P);
  conv_layer(ACT_P, ACT_Q, RW[0], RB[0], 2, 9, 128, 128, 65, 96, 64, 3, 1, 0, 1);
  conv_layer(ACT_Q, ACT_P, RW[1], RB[1], 2, 7, 128, 128, 64, 64, 64, 3, 1, 0, 1);
  conv_layer(ACT_P, ACT_Q, RW[2], RB[2], 2, 5, 128, 128, 64, 64, 64, 3, 1, 0, 1);
  conv_layer(ACT_Q, ACT_P, RW[3], RB[3], 2, 3, 128, 128, 64, 64, 64, 3, 1, 0, 1);
  conv_out1<<<dim3(cdivl(2L * 128 * 128, 256)), dim3(256), 0, stream>>>(ACT_P, ROW, ROB, HFRES2, 2, 128, 128, 64);

  // ---- hr = xIm[:,:,0] + hfRes + hfRes2; loss ------------------------------
  final_out<<<dim3(cdivl(32768, 256)), dim3(256), 0, stream>>>(xIm, HFRES, HFRES2, hrOut);
  lossA<<<dim3(338), dim3(256), 0, stream>>>(hrOut, target, LOSSS, 128, 128);
  lossB<<<dim3(1), dim3(32), 0, stream>>>(LOSSS, hrOut + 32768);
}
